// SeqSelfAttention_11879879542635
// MI455X (gfx1250) — matrix operand order repaired and checked by reading, not compiled, not run on hardware
//
#include <hip/hip_runtime.h>

typedef __attribute__((ext_vector_type(16))) _Float16 v16h;
typedef __attribute__((ext_vector_type(8)))  float    v8f;

constexpr int B_ = 4, L_ = 1024, D_ = 128, U_ = 32, W_ = 64;
constexpr int LW   = 96;    // padded key-window rows per 16-query tile (union is 79)
constexpr int XSTR = 130;   // f16 stride of x-window rows (odd dword count -> no LDS bank conflicts)
constexpr int WSTR = 130;   // f16 stride of transposed weight rows
constexpr int ASTR = 100;   // f16 stride of attention-probability rows

__global__ __launch_bounds__(128)
void seq_self_attn_kernel(const float* __restrict__ x,
                          const float* __restrict__ Wt,
                          const float* __restrict__ Wx,
                          const float* __restrict__ bh,
                          const float* __restrict__ Wa,
                          const float* __restrict__ ba,
                          float* __restrict__ out)
{
    // ---- LDS (65,280 bytes total) ----
    __shared__ _Float16 xw  [LW][XSTR];  // x window rows, f16           24,960 B
    __shared__ _Float16 wtT [U_][WSTR];  // Wt^T (u,k), f16               8,320 B
    __shared__ _Float16 wxT [U_][WSTR];  // Wx^T (u,k), f16               8,320 B
    __shared__ float    q32 [16][U_];    // q tile f32                    2,048 B
    __shared__ float    kwin[LW][U_];    // k window f32                 12,288 B
    __shared__ float    evals[16][LW];   // e then exp(e-m)               6,144 B
    __shared__ _Float16 aprob[16][ASTR]; // normalized probs f16          3,200 B

    const int tid  = threadIdx.x;
    const int lane = tid & 31;
    const int wave = tid >> 5;
    const int half = lane >> 4;   // which 16-lane half of the wave
    const int l16  = lane & 15;
    const int q0   = blockIdx.x * 16;   // first query row of this tile
    const int b    = blockIdx.y;        // batch
    const int j0   = q0 - W_ / 2;       // first key row of the (padded) window

    // ---------- phase 0: cooperative loads into LDS ----------
    for (int idx = tid; idx < LW * D_; idx += 128) {
        int r = idx >> 7, c = idx & 127;
        int j = j0 + r;
        float v = (j >= 0 && j < L_) ? x[((size_t)b * L_ + j) * D_ + c] : 0.f;
        xw[r][c] = (_Float16)v;
    }
    for (int idx = tid; idx < D_ * U_; idx += 128) {
        int k = idx >> 5, u = idx & 31;          // Wt is [D,U] row-major: idx = k*32+u
        wtT[u][k] = (_Float16)Wt[idx];
        wxT[u][k] = (_Float16)Wx[idx];
    }
    __syncthreads();

    // ---------- phase 1: q = x*Wt, k = x*Wx via WMMA ----------
    // 7 output tiles of 16 rows x 32 units: tile 0 -> q (A rows 32..47 of window),
    // tiles 1..6 -> k window rows. 2 N-tiles x 4 K-chunks = 8 WMMAs per tile.
    for (int t = wave; t < 7; t += 4) {
        const int arow0 = (t == 0) ? 32 : (t - 1) * 16;
        const _Float16 (*wT)[WSTR] = (t == 0) ? wtT : wxT;
        for (int nt = 0; nt < 2; ++nt) {
            v8f acc = {0.f, 0.f, 0.f, 0.f, 0.f, 0.f, 0.f, 0.f};
            for (int kc = 0; kc < 4; ++kc) {
                v16h af, bf;
                const _Float16* arow = &xw[arow0 + l16][0];
#pragma unroll
                for (int vg = 0; vg < 8; ++vg) {   // A 16x32 f16 layout (ISA 7.12.2)
                    int kk = kc * 32 + half * 8 + ((vg < 4) ? 2 * vg : 2 * vg + 8);
                    af[2 * vg]     = arow[kk];
                    af[2 * vg + 1] = arow[kk + 1];
                }
                const _Float16* brow = &wT[nt * 16 + l16][0];
#pragma unroll
                for (int vg = 0; vg < 8; ++vg) {   // B 32x16: col=lane&15, K same as A (table order)
                    int kk = kc * 32 + half * 8 + ((vg < 4) ? 2 * vg : 2 * vg + 8);
                    bf[2 * vg]     = brow[kk];
                    bf[2 * vg + 1] = brow[kk + 1];
                }
                acc = __builtin_amdgcn_wmma_f32_16x16x32_f16(
                        false, af, false, bf, (short)0, acc, false, false);
            }
            const int mb = half * 8;               // C/D: M = 8*half + vg, N = lane&15
#pragma unroll
            for (int vg = 0; vg < 8; ++vg) {
                if (t == 0) q32[mb + vg][nt * 16 + l16] = acc[vg];
                else        kwin[arow0 + mb + vg][nt * 16 + l16] = acc[vg];
            }
        }
    }
    __syncthreads();

    // ---------- phase 2: banded e[i,j] = Wa . tanh(q_i + k_j + bh) + ba ----------
    const float ba0 = ba[0];
    for (int idx = tid; idx < 16 * LW; idx += 128) {
        int qr = idx / LW, jj = idx - qr * LW;
        int j = j0 + jj;
        float e = -1e30f;
        // window: i-32 <= j < i+32  <=>  qr <= jj < qr+64
        if (j >= 0 && j < L_ && jj >= qr && jj < qr + W_) {
            float s = 0.f;
#pragma unroll 8
            for (int u = 0; u < U_; ++u)
                s += Wa[u] * tanhf(q32[qr][u] + kwin[jj][u] + bh[u]);
            e = s + ba0;
        }
        evals[qr][jj] = e;
    }
    __syncthreads();

    // ---------- phase 3: windowed softmax (max cancels except via eps) ----------
    if (tid < 16) {
        const int qr = tid;
        float m = -1e30f;
        for (int jj = 0; jj < LW; ++jj) m = fmaxf(m, evals[qr][jj]);
        float s = 0.f;
        for (int jj = 0; jj < LW; ++jj) {
            float ev = evals[qr][jj];
            float p  = (ev > -1e29f) ? __expf(ev - m) : 0.f;
            evals[qr][jj] = p;
            s += p;
        }
        const float inv = 1.f / (s + 1e-7f);
        for (int jj = 0; jj < LW; ++jj)
            aprob[qr][jj] = (_Float16)(evals[qr][jj] * inv);
    }
    __syncthreads();

    // ---------- phase 4: v = a @ x_window via WMMA (16x96 @ 96x128) ----------
    for (int nt = wave; nt < 8; nt += 4) {
        v8f acc = {0.f, 0.f, 0.f, 0.f, 0.f, 0.f, 0.f, 0.f};
        for (int kc = 0; kc < 3; ++kc) {
            v16h af, bf;
            const _Float16* arow = &aprob[l16][0];
#pragma unroll
            for (int vg = 0; vg < 8; ++vg) {
                int kk = kc * 32 + half * 8 + ((vg < 4) ? 2 * vg : 2 * vg + 8);
                af[2 * vg]     = arow[kk];
                af[2 * vg + 1] = arow[kk + 1];
            }
            const int col = nt * 16 + l16;
#pragma unroll
            for (int vg = 0; vg < 8; ++vg) {       // B = xw (K x N), K-pairs strided
                int K = kc * 32 + half * 8 + ((vg < 4) ? 2 * vg : 2 * vg + 8);
                bf[2 * vg]     = xw[K][col];
                bf[2 * vg + 1] = xw[K + 1][col];
            }
            acc = __builtin_amdgcn_wmma_f32_16x16x32_f16(
                    false, af, false, bf, (short)0, acc, false, false);
        }
        const int mb = half * 8;
#pragma unroll
        for (int vg = 0; vg < 8; ++vg)
            out[((size_t)b * L_ + q0 + mb + vg) * D_ + nt * 16 + l16] = acc[vg];
    }
}

extern "C" void kernel_launch(void* const* d_in, const int* in_sizes, int n_in,
                              void* d_out, int out_size, void* d_ws, size_t ws_size,
                              hipStream_t stream) {
    const float* x  = (const float*)d_in[0];
    const float* Wt = (const float*)d_in[1];
    const float* Wx = (const float*)d_in[2];
    const float* bh = (const float*)d_in[3];
    const float* Wa = (const float*)d_in[4];
    const float* ba = (const float*)d_in[5];
    float* out = (float*)d_out;
    (void)in_sizes; (void)n_in; (void)out_size; (void)d_ws; (void)ws_size;

    dim3 grid(L_ / 16, B_);   // 64 query tiles x 4 batches = 256 workgroups
    seq_self_attn_kernel<<<grid, 128, 0, stream>>>(x, Wt, Wx, bh, Wa, ba, out);
}
